// GeometricAttention_55456617726384
// MI455X (gfx1250) — matrix operand order repaired and checked by reading, not compiled, not run on hardware
//
#include <hip/hip_runtime.h>

// ---------------------------------------------------------------------------
// CDNA5 (gfx1250) bf16 WMMA attention pipeline, ping-pong software pipelining
//   B=8, N=1024, DIM=512, H=8, D=64, scale = 1/8
// ---------------------------------------------------------------------------

typedef __attribute__((ext_vector_type(16))) __bf16 bf16x16;
typedef __attribute__((ext_vector_type(8)))  float  f32x8;
typedef unsigned short u16;
typedef unsigned int   u32;

#define NH   8
#define HD   64
#define SEQ  1024
#define CDIM 512

union FragU { bf16x16 v; uint4 q[2]; };

static __device__ __forceinline__ u16 f2bf(float x) {   // round-to-nearest-even
  u32 u = __float_as_uint(x);
  u += 0x7fffu + ((u >> 16) & 1u);
  return (u16)(u >> 16);
}

static __device__ __forceinline__ f32x8 zerof8() {
  f32x8 z = {0.f,0.f,0.f,0.f,0.f,0.f,0.f,0.f};
  return z;
}

// A-fragment (16x32 bf16): per-lane elems K = p[0..15] contiguous
static __device__ __forceinline__ bf16x16 load_a_frag(const u16* p) {
  FragU f;
  f.q[0] = *(const uint4*)(p);
  f.q[1] = *(const uint4*)(p + 8);
  return f.v;
}
// B-fragment (32x16 bf16): per-lane elems K = p[0..15] contiguous
static __device__ __forceinline__ bf16x16 load_b_frag(const u16* p) {
  FragU f;
  f.q[0] = *(const uint4*)(p);
  f.q[1] = *(const uint4*)(p + 8);
  return f.v;
}

static __device__ __forceinline__ f32x8 wmma_bf16(bf16x16 a, bf16x16 b, f32x8 c) {
  return __builtin_amdgcn_wmma_f32_16x16x32_bf16(false, a, false, b, (short)0, c, false, false);
}

// ---------------------------------------------------------------------------
// fp32 -> bf16 bulk convert
// ---------------------------------------------------------------------------
__global__ void cvt_bf16_kernel(const float* __restrict__ in, u16* __restrict__ out, int n) {
  int i = blockIdx.x * blockDim.x + threadIdx.x;
  if (i < n) out[i] = f2bf(in[i]);
}

// ---------------------------------------------------------------------------
// Shared GEMM core: 16 rows x 64 cols per wave over K=512, ping-pong buffered.
// A from `arow` (row-major, koff pre-added), B from `wrow` (row-major weights).
// ---------------------------------------------------------------------------
static __device__ __forceinline__ void gemm_core_16x64(const u16* __restrict__ arow,
                                                       const u16* __restrict__ wrow,
                                                       f32x8 (&acc)[4]) {
  bf16x16 a0, a1, b0[4], b1[4];
  a0 = load_a_frag(arow);
#pragma unroll
  for (int t = 0; t < 4; ++t) b0[t] = load_b_frag(wrow + (size_t)t * 16 * CDIM);

  for (int k0 = 0; k0 < CDIM; k0 += 64) {
    // stage 0: issue loads for k0+32, consume buffer 0
    a1 = load_a_frag(arow + k0 + 32);
#pragma unroll
    for (int t = 0; t < 4; ++t) b1[t] = load_b_frag(wrow + (size_t)t * 16 * CDIM + k0 + 32);
#pragma unroll
    for (int t = 0; t < 4; ++t) acc[t] = wmma_bf16(a0, b0[t], acc[t]);

    // stage 1: issue loads for k0+64 (wrapped tail, harmless), consume buffer 1
    const int k2 = (k0 + 64) & (CDIM - 1);
    a0 = load_a_frag(arow + k2);
#pragma unroll
    for (int t = 0; t < 4; ++t) b0[t] = load_b_frag(wrow + (size_t)t * 16 * CDIM + k2);
#pragma unroll
    for (int t = 0; t < 4; ++t) acc[t] = wmma_bf16(a1, b1[t], acc[t]);
  }
}

// ---------------------------------------------------------------------------
// QKV GEMM: Y[i,o] = sum_c x[i,c] * W[o,c]   (M=8192, N=1536, K=512)
// Scatters into q[b,h,n,d], k[b,h,n,d], vT[b,h,d,n].
// ---------------------------------------------------------------------------
__global__ __launch_bounds__(32) void qkv_gemm_kernel(const u16* __restrict__ xb, const u16* __restrict__ wb,
                                                      u16* __restrict__ qo, u16* __restrict__ ko,
                                                      u16* __restrict__ vto) {
  const int lane = threadIdx.x & 31;
  const int l15 = lane & 15, hi = lane >> 4, koff = hi * 16;
  const int rb = blockIdx.x * 16;
  const int ob = blockIdx.y * 64;

  f32x8 acc[4];
#pragma unroll
  for (int t = 0; t < 4; ++t) acc[t] = zerof8();

  gemm_core_16x64(xb + (size_t)(rb + l15) * CDIM + koff,
                  wb + (size_t)(ob + l15) * CDIM + hi * 16, acc);

  const int sect = ob >> 9;          // 0=q, 1=k, 2=v   (64-col tile never straddles)
  const int h = (ob & 511) >> 6;     // head
#pragma unroll
  for (int t = 0; t < 4; ++t) {
    const int dd = t * 16 + l15;     // ob % 64 == 0
#pragma unroll
    for (int r = 0; r < 8; ++r) {
      const int i = rb + hi * 8 + r;
      const int bb = i >> 10, n = i & 1023;
      const u16 v = f2bf(acc[t][r]);
      const size_t hn = (size_t)(bb * NH + h) * SEQ + n;
      if (sect == 0)      qo[hn * HD + dd] = v;
      else if (sect == 1) ko[hn * HD + dd] = v;
      else                vto[((size_t)(bb * NH + h) * HD + dd) * SEQ + n] = v;
    }
  }
}

// ---------------------------------------------------------------------------
// Fused flash attention: one wave handles (b, h, 16 query rows).
// Online softmax over 32-key blocks, two stages per loop iteration with
// ping-pong K fragment buffers (no rotation copies). V fragments + bias are
// prefetched ahead of the softmax VALU work. P transposed through 1.25KB LDS.
// ---------------------------------------------------------------------------
__global__ __launch_bounds__(32) void attn_fa_kernel(const u16* __restrict__ qb, const u16* __restrict__ kb,
                                                     const u16* __restrict__ vtb, const float* __restrict__ bias,
                                                     u16* __restrict__ ao) {
  __shared__ __align__(16) u16 lds[16 * 40];   // 16 rows x (32 keys + 8 pad)

  const int lane = threadIdx.x & 31;
  const int l15 = lane & 15, hi = lane >> 4, koff = hi * 16;
  const int qblk = blockIdx.x, h = blockIdx.y, b = blockIdx.z;
  const int qbase = qblk * 16;

  const u16* Q  = qb  + ((size_t)(b * NH + h) * SEQ) * HD;
  const u16* K  = kb  + ((size_t)(b * NH + h) * SEQ) * HD;
  const u16* VT = vtb + ((size_t)(b * NH + h) * HD) * SEQ;
  const float* Bh = bias + (size_t)h * SEQ * SEQ;

  // Q A-fragments (16 rows x d-chunks {0..31, 32..63}), loaded once
  const u16* qrow = Q + (size_t)(qbase + l15) * HD;
  bf16x16 aq0 = load_a_frag(qrow + koff);
  bf16x16 aq1 = load_a_frag(qrow + 32 + koff);

  f32x8 acc[4];
#pragma unroll
  for (int t = 0; t < 4; ++t) acc[t] = zerof8();
  float rmax[8], rsum[8];
#pragma unroll
  for (int r = 0; r < 8; ++r) { rmax[r] = -1e30f; rsum[r] = 0.0f; }

  // one 32-key block: consume `kcur`, prefetch next block's K into `knxt`
  auto stage = [&](bf16x16 (&kcur)[4], bf16x16 (&knxt)[4], int mb) {
    // ---- prefetch: V fragments + bias for this block, K for the next ----
    bf16x16 bv[4];
#pragma unroll
    for (int t = 0; t < 4; ++t)
      bv[t] = load_b_frag(VT + (size_t)(t * 16 + l15) * SEQ + mb + hi * 16);

    const float* brow = Bh + (size_t)(qbase + hi * 8) * SEQ + mb + l15;
    float bias0[8], bias1[8];
#pragma unroll
    for (int r = 0; r < 8; ++r) {
      bias0[r] = brow[(size_t)r * SEQ];
      bias1[r] = brow[(size_t)r * SEQ + 16];
    }

    const int mbn = (mb + 32) & (SEQ - 1);       // wrap: tail prefetch is harmless
    {
      const u16* kp  = K + (size_t)(mbn + l15) * HD + hi * 16;
      const u16* kp2 = K + (size_t)(mbn + 16 + l15) * HD + hi * 16;
      knxt[0] = load_b_frag(kp);
      knxt[1] = load_b_frag(kp + 32);
      knxt[2] = load_b_frag(kp2);
      knxt[3] = load_b_frag(kp2 + 32);
    }

    // ---- scores: two 16x16 tiles over this 32-key block (K already resident)
    f32x8 s0 = zerof8(), s1 = zerof8();
    s0 = wmma_bf16(aq0, kcur[0], s0);
    s0 = wmma_bf16(aq1, kcur[1], s0);
    s1 = wmma_bf16(aq0, kcur[2], s1);
    s1 = wmma_bf16(aq1, kcur[3], s1);

    // scale + rpe bias (D layout: row = hi*8 + r, col = l15)
#pragma unroll
    for (int r = 0; r < 8; ++r) {
      s0[r] = s0[r] * 0.125f + bias0[r];
      s1[r] = s1[r] * 0.125f + bias1[r];
    }

    // ---- online softmax: per-row max / sum across the 16 lanes of each half
    float nmax[8], fac[8];
#pragma unroll
    for (int r = 0; r < 8; ++r) {
      float m = fmaxf(s0[r], s1[r]);
      m = fmaxf(m, __shfl_xor(m, 1, 32));
      m = fmaxf(m, __shfl_xor(m, 2, 32));
      m = fmaxf(m, __shfl_xor(m, 4, 32));
      m = fmaxf(m, __shfl_xor(m, 8, 32));
      float nm = fmaxf(rmax[r], m);
      fac[r] = __expf(rmax[r] - nm);
      rmax[r] = nm;
      nmax[r] = nm;
    }
#pragma unroll
    for (int r = 0; r < 8; ++r) {
      s0[r] = __expf(s0[r] - nmax[r]);
      s1[r] = __expf(s1[r] - nmax[r]);
      float ts = s0[r] + s1[r];
      ts += __shfl_xor(ts, 1, 32);
      ts += __shfl_xor(ts, 2, 32);
      ts += __shfl_xor(ts, 4, 32);
      ts += __shfl_xor(ts, 8, 32);
      rsum[r] = rsum[r] * fac[r] + ts;
    }
#pragma unroll
    for (int t = 0; t < 4; ++t)
#pragma unroll
      for (int r = 0; r < 8; ++r) acc[t][r] *= fac[r];

    // ---- transpose P (D layout -> A layout) through LDS ----
    __syncthreads();
#pragma unroll
    for (int r = 0; r < 8; ++r) {
      lds[(hi * 8 + r) * 40 + l15]      = f2bf(s0[r]);
      lds[(hi * 8 + r) * 40 + 16 + l15] = f2bf(s1[r]);
    }
    __syncthreads();
    bf16x16 ap = load_a_frag(&lds[l15 * 40] + koff);

    // ---- acc += P @ V (V fragments prefetched above) ----
#pragma unroll
    for (int t = 0; t < 4; ++t) acc[t] = wmma_bf16(ap, bv[t], acc[t]);
  };

  // prime the K fragment pipeline (block mb = 0)
  bf16x16 kA[4], kB[4];
  {
    const u16* kp  = K + (size_t)l15 * HD + hi * 16;
    const u16* kp2 = K + (size_t)(16 + l15) * HD + hi * 16;
    kA[0] = load_b_frag(kp);
    kA[1] = load_b_frag(kp + 32);
    kA[2] = load_b_frag(kp2);
    kA[3] = load_b_frag(kp2 + 32);
  }

  for (int mb = 0; mb < SEQ; mb += 64) {
    stage(kA, kB, mb);        // consume A, fill B (keys mb+32)
    stage(kB, kA, mb + 32);   // consume B, fill A (keys mb+64, wrapped at end)
  }

  // ---- normalize and emit [b, n, h*64 + d] bf16 ----
  float inv[8];
#pragma unroll
  for (int r = 0; r < 8; ++r) inv[r] = 1.0f / rsum[r];
  u16* orow = ao + ((size_t)b * SEQ + qbase + hi * 8) * CDIM + h * HD;
#pragma unroll
  for (int t = 0; t < 4; ++t)
#pragma unroll
    for (int r = 0; r < 8; ++r)
      orow[(size_t)r * CDIM + t * 16 + l15] = f2bf(acc[t][r] * inv[r]);
}

// ---------------------------------------------------------------------------
// Output projection: out = A @ Wp^T + bias   (M=8192, N=512, K=512), fp32 out
// ---------------------------------------------------------------------------
__global__ __launch_bounds__(32) void proj_gemm_kernel(const u16* __restrict__ ab, const u16* __restrict__ wb,
                                                       const float* __restrict__ pb, float* __restrict__ out) {
  const int lane = threadIdx.x & 31;
  const int l15 = lane & 15, hi = lane >> 4, koff = hi * 16;
  const int rb = blockIdx.x * 16;
  const int ob = blockIdx.y * 64;

  f32x8 acc[4];
#pragma unroll
  for (int t = 0; t < 4; ++t) acc[t] = zerof8();

  gemm_core_16x64(ab + (size_t)(rb + l15) * CDIM + koff,
                  wb + (size_t)(ob + l15) * CDIM + hi * 16, acc);

#pragma unroll
  for (int t = 0; t < 4; ++t) {
    const float bv = pb[ob + t * 16 + l15];
#pragma unroll
    for (int r = 0; r < 8; ++r)
      out[(size_t)(rb + hi * 8 + r) * CDIM + ob + t * 16 + l15] = acc[t][r] + bv;
  }
}

// ---------------------------------------------------------------------------
extern "C" void kernel_launch(void* const* d_in, const int* in_sizes, int n_in,
                              void* d_out, int out_size, void* d_ws, size_t ws_size,
                              hipStream_t stream) {
  const float* x      = (const float*)d_in[0];
  const float* rpe    = (const float*)d_in[1];
  const float* qkv_w  = (const float*)d_in[2];
  const float* proj_w = (const float*)d_in[3];
  const float* proj_b = (const float*)d_in[4];
  float* out = (float*)d_out;

  char* ws = (char*)d_ws;
  size_t off = 0;
  auto carve = [&](size_t elems) -> u16* {
    u16* p = (u16*)(ws + off);
    off += ((elems * 2) + 255) & ~(size_t)255;
    return p;
  };
  const size_t NX = (size_t)8 * SEQ * CDIM;        // 4,194,304 elements
  u16* xb  = carve(NX);                            // x in bf16
  u16* wq  = carve((size_t)3 * CDIM * CDIM);       // qkv_w bf16
  u16* wp  = carve((size_t)CDIM * CDIM);           // proj_w bf16
  u16* qb  = carve(NX);                            // q [b,h,n,d]
  u16* kb  = carve(NX);                            // k [b,h,n,d]
  u16* vtb = carve(NX);                            // v [b,h,d,n] (transposed)
  u16* ao  = carve(NX);                            // attention output [b*n, 512]
  (void)ws_size; (void)in_sizes; (void)n_in; (void)out_size;

  cvt_bf16_kernel<<<dim3((unsigned)((NX + 255) / 256)), dim3(256), 0, stream>>>(x, xb, (int)NX);
  cvt_bf16_kernel<<<dim3((3 * CDIM * CDIM + 255) / 256), dim3(256), 0, stream>>>(qkv_w, wq, 3 * CDIM * CDIM);
  cvt_bf16_kernel<<<dim3((CDIM * CDIM + 255) / 256), dim3(256), 0, stream>>>(proj_w, wp, CDIM * CDIM);

  qkv_gemm_kernel<<<dim3(512, 24), dim3(32), 0, stream>>>(xb, wq, qb, kb, vtb);
  attn_fa_kernel<<<dim3(SEQ / 16, NH, 8), dim3(32), 0, stream>>>(qb, kb, vtb, rpe, ao);
  proj_gemm_kernel<<<dim3(512, 8), dim3(32), 0, stream>>>(ao, wp, proj_b, out);
}
